// VectorQuantizer_20701742366919
// MI455X (gfx1250) — matrix operand order repaired and checked by reading, not compiled, not run on hardware
//
#include <hip/hip_runtime.h>
#include <math.h>

// ---------------- CDNA5 WMMA types ----------------
typedef __attribute__((ext_vector_type(16))) _Float16 v16h;
typedef __attribute__((ext_vector_type(8)))  float    v8f;

union Frag { v16h h; uint4 u[2]; };

__device__ __forceinline__ v8f wmma16(v16h a, v16h b, v8f c) {
  // D = A(16x32 f16) * B(32x16 f16) + C(16x16 f32)
  return __builtin_amdgcn_wmma_f32_16x16x32_f16(false, a, false, b, (short)0, c,
                                                false, false);
}

// A-fragment loader: row-major f16 tile with row stride `ldh` halves.
// lane<16: row=lane, K = {0..7, 16..23}+kc*32 ; lane>=16: row=lane-16, K={8..15,24..31}
__device__ __forceinline__ v16h loadA(const _Float16* tile, int ldh, int lane, int kc) {
  int arow = lane & 15, ksel = (lane >> 4) & 1;
  const _Float16* p = tile + arow * ldh + kc * 32 + ksel * 8;
  Frag f;
  f.u[0] = *(const uint4*)p;
  f.u[1] = *(const uint4*)(p + 16);
  return f.h;
}

// B-fragment loader from a packed [N][K] row-major f16 matrix (column n of B == row n).
// lane<16: col=lane, K={0..7,16..23} of chunk ; lane>=16: col=lane-16, K={8..15,24..31}
__device__ __forceinline__ v16h loadB(const _Float16* nk, int ldh, int nbase, int lane, int kc) {
  int arow = lane & 15, ksel = (lane >> 4) & 1;
  const _Float16* p = nk + (size_t)(nbase + arow) * ldh + kc * 32 + ksel * 8;
  Frag f;
  f.u[0] = *(const uint4*)p;
  f.u[1] = *(const uint4*)(p + 16);
  return f.h;
}

// ---------------- problem constants ----------------
#define BB 8
#define SS 6
#define LL 4096
#define WW 128
#define VV 2048

// ---------------- setup kernels ----------------

__global__ void k_norm_emb(const float* __restrict__ emb, _Float16* __restrict__ embh) {
  __shared__ float red[WW];
  int row = blockIdx.x, t = threadIdx.x;
  float v = emb[row * WW + t];
  red[t] = v * v;
  __syncthreads();
  for (int s = 64; s > 0; s >>= 1) {
    if (t < s) red[t] += red[t + s];
    __syncthreads();
  }
  float rn = 1.0f / fmaxf(sqrtf(red[0]), 1e-12f);
  embh[row * WW + t] = (_Float16)(v * rn);
}

// pack conv weights (levels 0..4): dst[lvl][cout][ks*128+cin] = cw[lvl][cout][cin][ks]
__global__ void k_pack_conv(const float* __restrict__ cw, _Float16* __restrict__ dst) {
  const int total = 5 * WW * 384;
  for (int i = blockIdx.x * 256 + threadIdx.x; i < total; i += gridDim.x * 256) {
    int k = i % 384;
    int ln = i / 384;           // lvl*128 + cout
    int ks = k >> 7, cin = k & 127;
    dst[i] = (_Float16)cw[(size_t)ln * WW * 3 + cin * 3 + ks];
  }
}

// pack mlp weights transposed: dst[lvl][n][k] = w[lvl][k][n]  (levels 0..4)
__global__ void k_pack_mlpT(const float* __restrict__ w, _Float16* __restrict__ dst) {
  const int total = 5 * WW * WW;
  for (int i = blockIdx.x * 256 + threadIdx.x; i < total; i += gridDim.x * 256) {
    int k = i & 127;
    int n = (i >> 7) & 127;
    int lvl = i >> 14;
    dst[i] = (_Float16)w[((size_t)lvl * WW + k) * WW + n];
  }
}

// ---------------- pooling ----------------
// dst[b,t,c] = mean over `chunk` of src[b*bstride + (t*chunk+j)*W + c]
__global__ void k_downg(const float* __restrict__ src, size_t bstride,
                        float* __restrict__ dst, int ts, int chunk) {
  int total = BB * ts * WW;
  int idx = blockIdx.x * 256 + threadIdx.x;
  if (idx >= total) return;
  int c = idx & 127;
  int t = (idx >> 7) % ts;
  int b = idx / (ts * WW);
  const float* s = src + (size_t)b * bstride + (size_t)t * chunk * WW + c;
  float acc = 0.0f;
  for (int j = 0; j < chunk; ++j) acc += s[(size_t)j * WW];
  dst[idx] = acc / (float)chunk;
}

// ---------------- bicubic upsample (A = -0.75) ----------------
__device__ __forceinline__ float bc_c1(float u) { return (1.25f * u - 2.25f) * u * u + 1.0f; }
__device__ __forceinline__ float bc_c2(float u) { return ((-0.75f * u + 3.75f) * u - 6.0f) * u + 3.0f; }

__global__ void k_up(const float* __restrict__ src, float* __restrict__ dst,
                     int Tin, int Tout) {
  int total = BB * Tout * 32;           // each thread: 4 channels
  int idx = blockIdx.x * 256 + threadIdx.x;
  if (idx >= total) return;
  int c4 = idx & 31;
  int t  = (idx >> 5) % Tout;
  int b  = idx / (32 * Tout);
  float scale = (float)Tin / (float)Tout;
  float x = ((float)t + 0.5f) * scale - 0.5f;
  float x0 = floorf(x);
  float u = x - x0;
  int ix = (int)x0;
  float w0 = bc_c2(u + 1.0f), w1 = bc_c1(u), w2 = bc_c1(1.0f - u), w3 = bc_c2(2.0f - u);
  int hi = Tin - 1;
  int i0 = min(max(ix - 1, 0), hi), i1 = min(max(ix, 0), hi);
  int i2 = min(max(ix + 1, 0), hi), i3 = min(max(ix + 2, 0), hi);
  const float4* sp = (const float4*)(src + (size_t)b * Tin * WW);
  float4 s0 = sp[i0 * 32 + c4], s1 = sp[i1 * 32 + c4];
  float4 s2 = sp[i2 * 32 + c4], s3 = sp[i3 * 32 + c4];
  float4 r;
  r.x = w0 * s0.x + w1 * s1.x + w2 * s2.x + w3 * s3.x;
  r.y = w0 * s0.y + w1 * s1.y + w2 * s2.y + w3 * s3.y;
  r.z = w0 * s0.z + w1 * s1.z + w2 * s2.z + w3 * s3.z;
  r.w = w0 * s0.w + w1 * s1.w + w2 * s2.w + w3 * s3.w;
  ((float4*)(dst + (size_t)b * Tout * WW))[t * 32 + c4] = r;
}

// ---------------- WMMA kernels ----------------

// residual -> l2norm -> scores vs codebook (f16 WMMA, f32 acc) -> argmax (first-max).
__global__ void k_score(const float* __restrict__ down, const float* __restrict__ up,
                        const _Float16* __restrict__ embh, int* __restrict__ ids, int M) {
  __shared__ _Float16 Ah[16 * WW];
  __shared__ float red[16][17];
  __shared__ float cval[16][128];
  __shared__ int   cidx[16][128];
  int tid = threadIdx.x;
  int r0 = blockIdx.x * 16;

  int rr = tid >> 4, cseg = (tid & 15) * 8;
  int row = r0 + rr;
  float vals[8];
  float ss = 0.0f;
  for (int j = 0; j < 8; ++j) {
    float v = 0.0f;
    if (row < M) {
      size_t g = (size_t)row * WW + cseg + j;
      v = down[g];
      if (up) v -= up[g];
    }
    vals[j] = v;
    ss += v * v;
  }
  red[rr][tid & 15] = ss;
  __syncthreads();
  if (tid < 16) {
    float s = 0.0f;
    for (int j = 0; j < 16; ++j) s += red[tid][j];
    red[tid][16] = 1.0f / fmaxf(sqrtf(s), 1e-12f);
  }
  __syncthreads();
  float rn = red[rr][16];
  for (int j = 0; j < 8; ++j) Ah[rr * WW + cseg + j] = (_Float16)(vals[j] * rn);
  __syncthreads();

  int lane = tid & 31, wave = tid >> 5;
  int arow = lane & 15, ksel = (lane >> 4) & 1;
  v16h a0 = loadA(Ah, WW, lane, 0), a1 = loadA(Ah, WW, lane, 1);
  v16h a2 = loadA(Ah, WW, lane, 2), a3 = loadA(Ah, WW, lane, 3);

  float best[8];
  int bidx[8];
  for (int v = 0; v < 8; ++v) { best[v] = -3.4e38f; bidx[v] = 0; }

  for (int j = 0; j < 16; ++j) {
    int tile = wave + j * 8;          // increasing n per lane -> strict '>' keeps first max
    int nbase = tile * 16;
    if (j < 15) __builtin_prefetch(embh + (size_t)(nbase + 128) * WW, 0, 1);
    // two independent accumulator chains -> overlapping WMMA issue
    v8f acc0 = {}, acc1 = {};
    acc0 = wmma16(a0, loadB(embh, WW, nbase, lane, 0), acc0);
    acc1 = wmma16(a1, loadB(embh, WW, nbase, lane, 1), acc1);
    acc0 = wmma16(a2, loadB(embh, WW, nbase, lane, 2), acc0);
    acc1 = wmma16(a3, loadB(embh, WW, nbase, lane, 3), acc1);
    v8f acc = acc0 + acc1;
    int n = nbase + arow;
    for (int v = 0; v < 8; ++v)
      if (acc[v] > best[v]) { best[v] = acc[v]; bidx[v] = n; }
  }
  for (int v = 0; v < 8; ++v) {
    int r2 = v + ksel * 8;
    int cc = wave * 16 + arow;
    cval[r2][cc] = best[v];
    cidx[r2][cc] = bidx[v];
  }
  __syncthreads();
  if (tid < 16) {
    int row2 = r0 + tid;
    if (row2 < M) {
      float bv = cval[tid][0];
      int bi = cidx[tid][0];
      for (int j = 1; j < 128; ++j) {
        float v = cval[tid][j];
        int ii = cidx[tid][j];
        if (v > bv || (v == bv && ii < bi)) { bv = v; bi = ii; }
      }
      ids[row2] = bi;
    }
  }
}

// conv1d(k=3,pad=1) as 16-token x K=384 im2row GEMM; h_mid = up + 0.5*(conv + cb)
__global__ void k_conv(const float* __restrict__ up, const _Float16* __restrict__ wct,
                       const float* __restrict__ cb, float* __restrict__ mid, int T) {
  __shared__ _Float16 Ah[16 * 384];
  int tid = threadIdx.x;
  int blkPerB = (T + 15) / 16;
  int b = blockIdx.x / blkPerB;
  int t0 = (blockIdx.x % blkPerB) * 16;
  const float* upb = up + (size_t)b * T * WW;

  for (int e = tid; e < 16 * 384; e += 256) {
    int r = e / 384, kk = e % 384;
    int ks = kk >> 7, cin = kk & 127;
    int t = t0 + r + ks - 1;
    float v = (t >= 0 && t < T) ? upb[(size_t)t * WW + cin] : 0.0f;
    Ah[e] = (_Float16)v;
  }
  __syncthreads();

  int lane = tid & 31, wave = tid >> 5;
  int arow = lane & 15, ksel = (lane >> 4) & 1;
  v8f acc0 = {}, acc1 = {};
  for (int kc = 0; kc < 12; kc += 2) {
    acc0 = wmma16(loadA(Ah, 384, lane, kc),     loadB(wct, 384, wave * 16, lane, kc),     acc0);
    acc1 = wmma16(loadA(Ah, 384, lane, kc + 1), loadB(wct, 384, wave * 16, lane, kc + 1), acc1);
  }
  v8f acc = acc0 + acc1;

  int n = wave * 16 + arow;
  float bias = cb[n];
  for (int v = 0; v < 8; ++v) {
    int t = t0 + v + ksel * 8;
    if (t < T) {
      size_t g = (size_t)(b * T + t) * WW + n;
      mid[g] = up[g] + 0.5f * (acc[v] + bias);
    }
  }
}

// out = h_mid + gelu_exact(h_mid@w1 + b1) @ w2 + b2   (two fused WMMA GEMMs)
__global__ void k_mlp(const float* __restrict__ mid, const _Float16* __restrict__ w1t,
                      const float* __restrict__ b1, const _Float16* __restrict__ w2t,
                      const float* __restrict__ b2, float* __restrict__ out, int M) {
  __shared__ _Float16 Ah[16 * WW];
  __shared__ _Float16 Gh[16 * WW];
  int tid = threadIdx.x;
  int r0 = blockIdx.x * 16;
  for (int e = tid; e < 16 * WW; e += 256) {
    int r = e >> 7, c = e & 127;
    float v = (r0 + r < M) ? mid[(size_t)(r0 + r) * WW + c] : 0.0f;
    Ah[e] = (_Float16)v;
  }
  __syncthreads();

  int lane = tid & 31, wave = tid >> 5;
  int arow = lane & 15, ksel = (lane >> 4) & 1;
  int n = wave * 16 + arow;

  v8f p0 = {}, p1 = {};
  p0 = wmma16(loadA(Ah, WW, lane, 0), loadB(w1t, WW, wave * 16, lane, 0), p0);
  p1 = wmma16(loadA(Ah, WW, lane, 1), loadB(w1t, WW, wave * 16, lane, 1), p1);
  p0 = wmma16(loadA(Ah, WW, lane, 2), loadB(w1t, WW, wave * 16, lane, 2), p0);
  p1 = wmma16(loadA(Ah, WW, lane, 3), loadB(w1t, WW, wave * 16, lane, 3), p1);
  v8f acc = p0 + p1;
  float bias = b1[n];
  for (int v = 0; v < 8; ++v) {
    int r = v + ksel * 8;
    float x = acc[v] + bias;
    float g = 0.5f * x * (1.0f + erff(x * 0.70710678118654752f));
    Gh[r * WW + n] = (_Float16)g;
  }
  __syncthreads();

  v8f q0 = {}, q1 = {};
  q0 = wmma16(loadA(Gh, WW, lane, 0), loadB(w2t, WW, wave * 16, lane, 0), q0);
  q1 = wmma16(loadA(Gh, WW, lane, 1), loadB(w2t, WW, wave * 16, lane, 1), q1);
  q0 = wmma16(loadA(Gh, WW, lane, 2), loadB(w2t, WW, wave * 16, lane, 2), q0);
  q1 = wmma16(loadA(Gh, WW, lane, 3), loadB(w2t, WW, wave * 16, lane, 3), q1);
  v8f acc2 = q0 + q1;
  float bias2 = b2[n];
  for (int v = 0; v < 8; ++v) {
    int grow = r0 + v + ksel * 8;
    if (grow < M) {
      size_t g = (size_t)grow * WW + n;
      out[g] = mid[g] + acc2[v] + bias2;
    }
  }
}

// ---------------- decode helpers / losses ----------------

__global__ void k_gather_add(const float* __restrict__ emb, const int* __restrict__ ids,
                             float* __restrict__ dst, int M, int doAdd) {
  int idx = blockIdx.x * 256 + threadIdx.x;
  if (idx >= M * WW) return;
  int c = idx & 127, r = idx >> 7;
  float v = emb[(size_t)ids[r] * WW + c];
  dst[idx] = doAdd ? (dst[idx] + v) : v;
}

// deterministic MSE: fixed-grid partial sums (no float atomics)
__global__ void k_mse_part(const float* __restrict__ a, const float* __restrict__ b,
                           float* __restrict__ parts, int N) {
  __shared__ float red[256];
  float s = 0.0f;
  for (int i = blockIdx.x * 256 + threadIdx.x; i < N; i += gridDim.x * 256) {
    float d = a[i] - b[i];
    s += d * d;
  }
  red[threadIdx.x] = s;
  __syncthreads();
  for (int k = 128; k > 0; k >>= 1) {
    if (threadIdx.x < k) red[threadIdx.x] += red[threadIdx.x + k];
    __syncthreads();
  }
  if (threadIdx.x == 0) parts[blockIdx.x] = red[0];
}

__global__ void k_reduce_part(const float* __restrict__ parts, int count,
                              float* __restrict__ slot) {
  if (threadIdx.x == 0 && blockIdx.x == 0) {
    float s = 0.0f;
    for (int i = 0; i < count; ++i) s += parts[i];
    slot[0] = s;
  }
}

__global__ void k_loss(const float* __restrict__ accum, float* __restrict__ outp) {
  if (threadIdx.x == 0 && blockIdx.x == 0) {
    const float cnt[6] = {1024.f, 4096.f, 16384.f, 65536.f, 262144.f, 4194304.f};
    float commit = 0.0f;
    for (int i = 0; i < 6; ++i) commit += accum[i] / cnt[i];
    float codebook = accum[6] / 4194304.0f;
    outp[0] = codebook + 0.1f * commit / 6.0f;
  }
}

// out = f_mean + (f_hat - f_mean)   (faithful fp reproduction)
__global__ void k_final(const float* __restrict__ f, const float* __restrict__ fhat,
                        float* __restrict__ out) {
  int idx = blockIdx.x * 256 + threadIdx.x;
  if (idx >= BB * LL * WW) return;
  int c = idx & 127;
  int l = (idx >> 7) & (LL - 1);
  int b = idx >> 19;
  float s = 0.0f;
  for (int sidx = 0; sidx < SS; ++sidx)
    s += f[(((size_t)b * SS + sidx) * LL + l) * WW + c];
  float fm = s / 6.0f;
  out[idx] = fm + (fhat[idx] - fm);
}

// ---------------- launcher ----------------
extern "C" void kernel_launch(void* const* d_in, const int* in_sizes, int n_in,
                              void* d_out, int out_size, void* d_ws, size_t ws_size,
                              hipStream_t stream) {
  (void)in_sizes; (void)n_in; (void)out_size; (void)ws_size;
  const float* f   = (const float*)d_in[0];
  const float* emb = (const float*)d_in[1];
  const float* cw  = (const float*)d_in[2];
  const float* cb  = (const float*)d_in[3];
  const float* w1  = (const float*)d_in[4];
  const float* b1  = (const float*)d_in[5];
  const float* w2  = (const float*)d_in[6];
  const float* b2  = (const float*)d_in[7];
  float* out = (float*)d_out;

  const int SC[6] = {1, 4, 16, 64, 256, 4096};
  const int IDOFF[6] = {0, 8, 40, 168, 680, 2728};

  char* base = (char*)d_ws;
  size_t off = 0;
  auto alloc = [&](size_t bytes) {
    void* p = base + off;
    off = (off + bytes + 255) & ~(size_t)255;
    return p;
  };
  _Float16* embh = (_Float16*)alloc((size_t)VV * WW * 2);
  _Float16* wct  = (_Float16*)alloc((size_t)5 * WW * 384 * 2);
  _Float16* w1t  = (_Float16*)alloc((size_t)5 * WW * WW * 2);
  _Float16* w2t  = (_Float16*)alloc((size_t)5 * WW * WW * 2);
  int* ids       = (int*)alloc(35496 * 4);
  float* down[6];
  for (int i = 0; i < 6; ++i) down[i] = (float*)alloc((size_t)BB * SC[i] * WW * 4);
  float* X     = (float*)alloc((size_t)BB * LL * WW * 4);   // fhat / phi output (up_h)
  float* Y     = (float*)alloc((size_t)BB * LL * WW * 4);   // bicubic output / pool temp
  float* mid   = (float*)alloc((size_t)BB * LL * WW * 4);   // h_mid
  float* parts = (float*)alloc(1024 * 4);
  float* accum = (float*)alloc(8 * 4);

  auto cdiv = [](int a, int b) { return (a + b - 1) / b; };
  auto mse = [&](const float* a, const float* bb, int N, int slot) {
    int g = cdiv(N, 4096);
    if (g < 1) g = 1;
    if (g > 1024) g = 1024;
    k_mse_part<<<g, 256, 0, stream>>>(a, bb, parts, N);
    k_reduce_part<<<1, 32, 0, stream>>>(parts, g, accum + slot);
  };

  // setup
  k_norm_emb<<<VV, WW, 0, stream>>>(emb, embh);
  k_pack_conv<<<cdiv(5 * WW * 384, 256), 256, 0, stream>>>(cw, wct);
  k_pack_mlpT<<<cdiv(5 * WW * WW, 256), 256, 0, stream>>>(w1, w1t);
  k_pack_mlpT<<<cdiv(5 * WW * WW, 256), 256, 0, stream>>>(w2, w2t);

  // ---- Phase 1: encode ----
  for (int i = 0; i < 6; ++i) {
    int ts = SC[i];
    int s = (i == 0 || i == 5) ? 0 : (6 - i);
    int chunk = LL / ts;
    const float* slice = f + (size_t)s * LL * WW;
    if (chunk >= 256) {
      // two-stage pooling for parallelism: chunk = (chunk/64) * 64
      int a = chunk / 64;
      k_downg<<<cdiv(BB * ts * 64 * WW, 256), 256, 0, stream>>>(
          slice, (size_t)SS * LL * WW, Y, ts * 64, a);
      k_downg<<<cdiv(BB * ts * WW, 256), 256, 0, stream>>>(
          Y, (size_t)ts * 64 * WW, down[i], ts, 64);
    } else {
      k_downg<<<cdiv(BB * ts * WW, 256), 256, 0, stream>>>(
          slice, (size_t)SS * LL * WW, down[i], ts, chunk);
    }
    int M = BB * ts;
    k_score<<<cdiv(M, 16), 256, 0, stream>>>(down[i], (i == 0) ? nullptr : X,
                                             embh, ids + IDOFF[i], M);
    if (i < 5) {
      int tn = SC[i + 1];
      k_up<<<cdiv(BB * tn * 32, 256), 256, 0, stream>>>(down[i], Y, ts, tn);
      k_conv<<<BB * cdiv(tn, 16), 256, 0, stream>>>(Y, wct + (size_t)i * WW * 384,
                                                    cb + i * WW, mid, tn);
      k_mlp<<<cdiv(BB * tn, 16), 256, 0, stream>>>(mid, w1t + (size_t)i * WW * WW,
                                                   b1 + i * WW, w2t + (size_t)i * WW * WW,
                                                   b2 + i * WW, X, BB * tn);
    }
  }

  // ---- Phase 2: decode + losses ----
  k_gather_add<<<cdiv(BB * 1 * WW, 256), 256, 0, stream>>>(emb, ids + IDOFF[0], X, BB, 0);
  mse(down[0], X, BB * 1 * WW, 0);
  for (int i = 1; i < 6; ++i) {
    int ts = SC[i];
    int lvl = i - 1;
    k_up<<<cdiv(BB * ts * 32, 256), 256, 0, stream>>>(X, Y, SC[i - 1], ts);
    k_conv<<<BB * cdiv(ts, 16), 256, 0, stream>>>(Y, wct + (size_t)lvl * WW * 384,
                                                  cb + lvl * WW, mid, ts);
    k_mlp<<<cdiv(BB * ts, 16), 256, 0, stream>>>(mid, w1t + (size_t)lvl * WW * WW,
                                                 b1 + lvl * WW, w2t + (size_t)lvl * WW * WW,
                                                 b2 + lvl * WW, X, BB * ts);
    k_gather_add<<<cdiv(BB * ts * WW, 256), 256, 0, stream>>>(emb, ids + IDOFF[i], X,
                                                              BB * ts, 1);
    mse(down[i], X, BB * ts * WW, i);
  }
  mse(X, down[5], BB * LL * WW, 6);
  k_loss<<<1, 32, 0, stream>>>(accum, out + (size_t)BB * LL * WW);
  k_final<<<cdiv(BB * LL * WW, 256), 256, 0, stream>>>(f, X, out);
}
